// DCNv4Backbone_82205674045879
// MI455X (gfx1250) — compile-verified, matrix operand order checked by reading, not run on hardware
//
#include <hip/hip_runtime.h>
#include <hip/hip_bf16.h>
#include <math.h>

// ---------------------------------------------------------------------------
// Types for CDNA5 WMMA (wave32)
// ---------------------------------------------------------------------------
typedef __attribute__((ext_vector_type(16))) _Float16 v16h;
typedef __attribute__((ext_vector_type(8)))  _Float16 v8h;
typedef __attribute__((ext_vector_type(8)))  float    v8f;

#define BN_RSQ 0.9999950000374997f   // 1/sqrt(1 + 1e-5)

// ---------------------------------------------------------------------------
// Workspace arena layout (bytes)
// ---------------------------------------------------------------------------
static constexpr size_t OFF_ACT0  = 0;                         // 8*64*112*112*4
static constexpr size_t OFF_ACT1  = OFF_ACT0  + 25690112;      // 8*128*56*56*4
static constexpr size_t OFF_SEQ   = OFF_ACT1  + 12845056;      // 25088*128*2
static constexpr size_t OFF_V     = OFF_SEQ   + 6422528;       // 25088*128*4
static constexpr size_t OFF_OM    = OFF_V     + 12845056;      // 25088*112*4
static constexpr size_t OFF_SEQ2  = OFF_OM    + 11239424;      // 25088*128*2
static constexpr size_t OFF_SEQIN = OFF_SEQ2  + 6422528;       // 25088*64*2
static constexpr size_t OFF_W     = OFF_SEQIN + 3211264;       // packed weights

// ---------------------------------------------------------------------------
// Stem: 7x7 s2 p3 conv (3->64) + BN + exact GELU, NCHW f32
// ---------------------------------------------------------------------------
__global__ void k_stem(const float* __restrict__ x, const float* __restrict__ w,
                       const float* __restrict__ g, const float* __restrict__ b,
                       float* __restrict__ out)
{
    const int HO = 112, WO = 112, CO = 64, HI = 224, WI = 224, CI = 3;
    int i = blockIdx.x * 256 + threadIdx.x;
    int total = 8 * CO * HO * WO;
    if (i >= total) return;
    int xw = i % WO; int t = i / WO;
    int y  = t % HO; t /= HO;
    int co = t % CO; int img = t / CO;
    float s = 0.f;
    for (int ci = 0; ci < CI; ++ci)
        for (int ky = 0; ky < 7; ++ky) {
            int iy = y * 2 - 3 + ky;
            if (iy < 0 || iy >= HI) continue;
            for (int kx = 0; kx < 7; ++kx) {
                int ix = xw * 2 - 3 + kx;
                if (ix < 0 || ix >= WI) continue;
                s += x[(((size_t)img * CI + ci) * HI + iy) * WI + ix] *
                     w[((co * CI + ci) * 7 + ky) * 7 + kx];
            }
        }
    s = s * (g[co] * BN_RSQ) + b[co];
    out[i] = 0.5f * s * (1.0f + erff(s * 0.70710678118f));
}

// ---------------------------------------------------------------------------
// NCHW f32 -> (N*L) x C row-major f16 sequence (spatial stride folded in)
// ---------------------------------------------------------------------------
__global__ void k_pack_seq(const float* __restrict__ x, _Float16* __restrict__ seq,
                           int C, int Hin, int Win, int stride, int W, int L, int total)
{
    int i = blockIdx.x * 256 + threadIdx.x;
    if (i >= total) return;
    int c = i % C; int m = i / C;
    int img = m / L; int pix = m % L;
    int y = pix / W, xw = pix % W;
    seq[i] = (_Float16)x[(((size_t)img * C + c) * Hin + y * stride) * Win + xw * stride];
}

// ---------------------------------------------------------------------------
// Weight pack: f32 -> f16 B-transposed [Npad][K], zero-filled rows >= Nn.
//   trans==0: src is [K][Nn] (seq @ W layout) ; trans==1: src is [Nn][K]
// ---------------------------------------------------------------------------
__global__ void k_pack_wt(const float* __restrict__ w, _Float16* __restrict__ wt,
                          int K, int Nn, int Npad, int trans)
{
    int i = blockIdx.x * 256 + threadIdx.x;
    if (i >= Npad * K) return;
    int n = i / K, k = i % K;
    float val = 0.f;
    if (n < Nn) val = trans ? w[(size_t)n * K + k] : w[(size_t)k * Nn + n];
    wt[i] = (_Float16)val;
}

// ---------------------------------------------------------------------------
// Load 16 halves in the WMMA A/B per-lane pattern: 8 @ p, 8 @ p+16
// ---------------------------------------------------------------------------
__device__ __forceinline__ v16h ld16(const _Float16* p)
{
    v8h lo = *(const v8h*)(p);
    v8h hi = *(const v8h*)(p + 16);
    v16h r;
#pragma unroll
    for (int i = 0; i < 8; ++i) { r[i] = lo[i]; r[i + 8] = hi[i]; }
    return r;
}

// ---------------------------------------------------------------------------
// WMMA GEMM epilogue store. C/D layout: lane 0-15 -> rows m0..m0+7, col=lane;
// lane 16-31 -> rows m0+8..m0+15, col=lane-16. Columns >= Nn masked out.
// ---------------------------------------------------------------------------
__device__ __forceinline__ void store_tile(
    v8f acc, int m0, int n, int lane, int Nn, int L,
    const float* __restrict__ bias, const float* __restrict__ bng,
    const float* __restrict__ bnb,  const float* __restrict__ resid,
    float* __restrict__ out, int mode)
{
    if (n >= Nn) return;                 // only possible in padded (om) GEMM
    int mbase = m0 + ((lane < 16) ? 0 : 8);
#pragma unroll
    for (int i = 0; i < 8; ++i) {
        int m = mbase + i;
        float t = acc[i];
        if (mode == 0) {
            out[(size_t)m * Nn + n] = t + bias[n];
        } else {
            int img = m / L, pix = m % L;
            size_t adr = ((size_t)img * Nn + n) * L + pix;   // NCHW, C == Nn
            if (mode == 1) {
                out[adr] = t * (bng[n] * BN_RSQ) + bnb[n];
            } else {
                t += bias[n];
                t = t * (bng[n] * BN_RSQ) + bnb[n];
                float ge = 0.5f * t * (1.0f + erff(t * 0.70710678118f));
                out[adr] = ge + resid[adr];
            }
        }
    }
}

// ---------------------------------------------------------------------------
// WMMA GEMM: out = A[M][K](f16) x Bt[Npad][K](f16)^T  (+ fused epilogue)
// 128 threads = 4 waves; workgroup tile 16 x 128 (2 N-tiles/wave, all valid
// by construction -> no EXEC masking around WMMA). A strip staged in LDS.
// K-loop is register double-buffered so next chunk's b128 loads overlap the
// current pair of v_wmma_f32_16x16x32_f16.
// ---------------------------------------------------------------------------
__global__ __launch_bounds__(128) void k_gemm(
    const _Float16* __restrict__ A, const _Float16* __restrict__ Bt,
    const float* __restrict__ bias, const float* __restrict__ bng,
    const float* __restrict__ bnb,  const float* __restrict__ resid,
    float* __restrict__ out, int M, int Nn, int K, int L, int mode)
{
    __shared__ __align__(16) _Float16 As[16 * 512];   // 16 KB max (K<=512)
    const int tid = threadIdx.x;
    const int m0  = blockIdx.x << 4;

    { // cooperative 16B-chunk stage of the 16xK A strip
        const uint4* src = (const uint4*)(A + (size_t)m0 * K);
        uint4* dst = (uint4*)As;
        const int nch = 2 * K;          // 16*K halves / 8 per chunk
        for (int i = tid; i < nch; i += 128) dst[i] = src[i];
    }
    __syncthreads();

    const int wave = tid >> 5;
    const int lane = tid & 31;
    const int lm   = lane & 15;
    const int lk   = (lane >> 4) << 3;  // K sub-block: 0 or 8 per lane half

    const int nt0 = (blockIdx.y * 8 + wave * 2) << 4;   // always < Npad
    const int nt1 = nt0 + 16;

    const _Float16* ap  = As + lm * K + lk;
    const _Float16* bq0 = Bt + (size_t)(nt0 + lm) * K + lk;
    const _Float16* bq1 = Bt + (size_t)(nt1 + lm) * K + lk;

    v8f acc0 = {}; v8f acc1 = {};
    v16h a_c  = ld16(ap);
    v16h b0_c = ld16(bq0);
    v16h b1_c = ld16(bq1);
    for (int k0 = 32; k0 < K; k0 += 32) {
        __builtin_prefetch((const void*)(bq0 + k0 + 96), 0, 1);
        v16h a_n  = ld16(ap  + k0);
        v16h b0_n = ld16(bq0 + k0);
        v16h b1_n = ld16(bq1 + k0);
        acc0 = __builtin_amdgcn_wmma_f32_16x16x32_f16(false, a_c, false, b0_c,
                                                      (short)0, acc0, false, false);
        acc1 = __builtin_amdgcn_wmma_f32_16x16x32_f16(false, a_c, false, b1_c,
                                                      (short)0, acc1, false, false);
        a_c = a_n; b0_c = b0_n; b1_c = b1_n;
    }
    acc0 = __builtin_amdgcn_wmma_f32_16x16x32_f16(false, a_c, false, b0_c,
                                                  (short)0, acc0, false, false);
    acc1 = __builtin_amdgcn_wmma_f32_16x16x32_f16(false, a_c, false, b1_c,
                                                  (short)0, acc1, false, false);

    store_tile(acc0, m0, nt0 + lm, lane, Nn, L, bias, bng, bnb, resid, out, mode);
    store_tile(acc1, m0, nt1 + lm, lane, Nn, L, bias, bng, bnb, resid, out, mode);
}

// ---------------------------------------------------------------------------
// DCNv4 deformable aggregation: one wave per (pixel row, group).
// om row layout: per g (27 of 112): [off_x,off_y]*9 then mask*9.
// Output f16 sequence for the op GEMM.
// ---------------------------------------------------------------------------
__global__ __launch_bounds__(128) void k_dcn(
    const float* __restrict__ v, const float* __restrict__ om,
    _Float16* __restrict__ outseq, int L, int H, int W, int C)
{
    const int gw   = blockIdx.x * 4 + (threadIdx.x >> 5);
    const int lane = threadIdx.x & 31;
    const int row  = gw >> 2;
    const int g    = gw & 3;
    const int img  = row / L;
    const int l    = row % L;
    const int Cg   = C >> 2;
    const int nchunk = Cg >> 5;          // 1..4
    const int gbase  = g * Cg;

    const float* omr = om + (size_t)row * 112 + g * 27;
    const float refx = (float)(l % W);
    const float refy = (float)(l / W);

    float acc[4] = {0.f, 0.f, 0.f, 0.f};
    const float* vimg = v + (size_t)img * L * C + gbase + lane;

    for (int p = 0; p < 9; ++p) {
        float ox = omr[2 * p], oy = omr[2 * p + 1], mk = omr[18 + p];
        float px = refx + (float)(p % 3 - 1) + ox;
        float py = refy + (float)(p / 3 - 1) + oy;
        float x0 = floorf(px), y0 = floorf(py);
        float wx1 = px - x0, wx0 = 1.f - wx1;
        float wy1 = py - y0, wy0 = 1.f - wy1;
#pragma unroll
        for (int cy = 0; cy < 2; ++cy)
#pragma unroll
            for (int cx = 0; cx < 2; ++cx) {
                float xi = x0 + (float)cx, yi = y0 + (float)cy;
                if (xi < 0.f || xi > (float)(W - 1) ||
                    yi < 0.f || yi > (float)(H - 1)) continue;  // zero weight
                float wgt = (cy ? wy1 : wy0) * (cx ? wx1 : wx0) * mk;
                int idx = (int)yi * W + (int)xi;
                const float* vr = vimg + (size_t)idx * C;
                for (int cc = 0; cc < nchunk; ++cc)
                    acc[cc] += wgt * vr[cc * 32];
            }
    }
    _Float16* o = outseq + (size_t)row * C + gbase + lane;
    for (int cc = 0; cc < nchunk; ++cc) o[cc * 32] = (_Float16)acc[cc];
}

// ---------------------------------------------------------------------------
// Host orchestration
// ---------------------------------------------------------------------------
struct BP {
    const float *proj_w, *proj_g, *proj_b;
    const float *vp_w, *vp_b, *om_w, *om_b, *op_w, *op_b, *bn_g, *bn_b;
};

static void run_block(const BP& P, const float* in, float* xpbuf, float* out,
                      int Nimg, int Cin, int C, int Hin, int Win, int stride,
                      char* ws, hipStream_t stream)
{
    const int H = Hin / stride, W = Win / stride;
    const int L = H * W, M = Nimg * L;

    _Float16* seq   = (_Float16*)(ws + OFF_SEQ);
    float*    vbuf  = (float*)   (ws + OFF_V);
    float*    ombuf = (float*)   (ws + OFF_OM);
    _Float16* seq2  = (_Float16*)(ws + OFF_SEQ2);
    _Float16* seqin = (_Float16*)(ws + OFF_SEQIN);
    _Float16* wt_proj = (_Float16*)(ws + OFF_W);
    _Float16* wt_vp   = (_Float16*)(ws + OFF_W + 0x80000);
    _Float16* wt_om   = (_Float16*)(ws + OFF_W + 0x100000);
    _Float16* wt_op   = (_Float16*)(ws + OFF_W + 0x120000);

    const float* xp;
    if (P.proj_w) {
        int nw = C * Cin;   // Npad == C (multiple of 128)
        k_pack_wt<<<(nw + 255) / 256, 256, 0, stream>>>(P.proj_w, wt_proj,
                                                        Cin, C, C, 1);
        int ns = M * Cin;
        k_pack_seq<<<(ns + 255) / 256, 256, 0, stream>>>(in, seqin, Cin, Hin, Win,
                                                         stride, W, L, ns);
        k_gemm<<<dim3(M / 16, C / 128), 128, 0, stream>>>(
            seqin, wt_proj, nullptr, P.proj_g, P.proj_b, nullptr,
            xpbuf, M, C, Cin, L, 1);
        xp = xpbuf;
    } else {
        xp = in;
    }
    {
        int ns = M * C;
        k_pack_seq<<<(ns + 255) / 256, 256, 0, stream>>>(xp, seq, C, H, W, 1, W, L, ns);
    }
    k_pack_wt<<<(C * C + 255) / 256, 256, 0, stream>>>(P.vp_w, wt_vp, C, C, C, 0);
    k_gemm<<<dim3(M / 16, C / 128), 128, 0, stream>>>(
        seq, wt_vp, P.vp_b, nullptr, nullptr, nullptr, vbuf, M, C, C, L, 0);
    // om: Nn=112 padded to 128 (zero rows in Bt, store-masked columns)
    k_pack_wt<<<(C * 128 + 255) / 256, 256, 0, stream>>>(P.om_w, wt_om, C, 112, 128, 0);
    k_gemm<<<dim3(M / 16, 1), 128, 0, stream>>>(
        seq, wt_om, P.om_b, nullptr, nullptr, nullptr, ombuf, M, 112, C, L, 0);
    k_dcn<<<M, 128, 0, stream>>>(vbuf, ombuf, seq2, L, H, W, C);
    k_pack_wt<<<(C * C + 255) / 256, 256, 0, stream>>>(P.op_w, wt_op, C, C, C, 0);
    k_gemm<<<dim3(M / 16, C / 128), 128, 0, stream>>>(
        seq2, wt_op, P.op_b, P.bn_g, P.bn_b, xp, out, M, C, C, L, 2);
}

extern "C" void kernel_launch(void* const* d_in, const int* in_sizes, int n_in,
                              void* d_out, int out_size, void* d_ws, size_t ws_size,
                              hipStream_t stream)
{
    (void)in_sizes; (void)n_in; (void)out_size; (void)ws_size;
    char* ws = (char*)d_ws;
    float* act0 = (float*)(ws + OFF_ACT0);
    float* act1 = (float*)(ws + OFF_ACT1);

    auto F = [&](int i) { return (const float*)d_in[i]; };
    const float* x      = F(0);
    const float* stem_w = F(1);
    const float* stem_g = F(2);
    const float* stem_b = F(3);
    BP p1{F(4),  F(5),  F(6),  F(7),  F(8),  F(9),  F(10), F(11), F(12), F(13), F(14)};
    BP p2{F(15), F(16), F(17), F(18), F(19), F(20), F(21), F(22), F(23), F(24), F(25)};
    BP p3{F(26), F(27), F(28), F(29), F(30), F(31), F(32), F(33), F(34), F(35), F(36)};
    BP p4{nullptr, nullptr, nullptr,
          F(37), F(38), F(39), F(40), F(41), F(42), F(43), F(44)};

    // Stem conv + BN + GELU -> act0 (8x64x112x112)
    {
        int total = 8 * 64 * 112 * 112;
        k_stem<<<(total + 255) / 256, 256, 0, stream>>>(x, stem_w, stem_g, stem_b, act0);
    }
    run_block(p1, act0, act1, act0,        8,  64, 128, 112, 112, 2, ws, stream);
    run_block(p2, act0, act1, act0,        8, 128, 256,  56,  56, 2, ws, stream);
    run_block(p3, act0, act1, act0,        8, 256, 512,  28,  28, 2, ws, stream);
    run_block(p4, act0, act1, (float*)d_out, 8, 512, 512, 14, 14, 1, ws, stream);
}
